// EmbeddingLayer_80616536146205
// MI455X (gfx1250) — 8-bit matrix operand order repaired and checked by reading, compiled, not run on hardware
//
#include <hip/hip_runtime.h>
#include <hip/hip_bf16.h>
#include <math.h>

#define Bn      64
#define Cc      3
#define Hh      224
#define Ww      224
#define Pp      7
#define GH      32
#define GW      32
#define NTOK    1024
#define PD      147
#define KPAD    192
#define Dd      1024
#define EPS     1e-5f

typedef __attribute__((ext_vector_type(8))) int v8i;

__global__ void wmean_kernel(const float* __restrict__ W, float* __restrict__ sw_inv) {
    __shared__ float red[256];
    const int tid = threadIdx.x;
    float s = 0.0f;
    const int total = Dd * PD;
    for (int i = tid; i < total; i += 256) s += fabsf(W[i]);
    red[tid] = s;
    __syncthreads();
    for (int off = 128; off > 0; off >>= 1) {
        if (tid < off) red[tid] += red[tid + off];
        __syncthreads();
    }
    if (tid == 0) sw_inv[0] = fmaxf(red[0] * (1.0f / (float)total), 1e-5f);
}

__global__ void wquant_kernel(const float* __restrict__ W,
                              const float* __restrict__ sw_inv_p,
                              signed char* __restrict__ wq) {
    int idx = blockIdx.x * 256 + threadIdx.x;
    if (idx >= Dd * KPAD) return;
    int d = idx / KPAD;
    int k = idx - d * KPAD;
    float swi = sw_inv_p[0];
    float sw  = 1.0f / swi;
    float v = (k < PD) ? W[d * PD + k] : 0.0f;
    int q = (int)rintf(v * sw);
    q = q > 1 ? 1 : (q < -1 ? -1 : q);
    int ntile = d >> 4, nn = d & 15;
    int kb = k >> 6, kk = k & 63;
    int seg = kk >> 5, rem = kk & 31, h = rem >> 4, r2 = rem & 15;
    int p = seg * 4 + (r2 >> 2), byt = r2 & 3;
    int lane = nn + 16 * h;
    size_t addr = ((size_t)((ntile * 3 + kb) * 32 + lane)) * 32 + p * 4 + byt;
    wq[addr] = (signed char)q;
}

__global__ void posemb_kernel(float* __restrict__ pe) {
    const int n = blockIdx.x;
    const int gh = n >> 5, gw = n & 31;
    const float lnT = logf(10000.0f);
    for (int d = threadIdx.x; d < Dd; d += 256) {
        int seg = d >> 8;
        int i = d & 255;
        float omega = expf(-((float)i * (1.0f / 255.0f)) * lnT);
        float val;
        if (seg == 0)      val = sinf((float)gw * omega);
        else if (seg == 1) val = cosf((float)gw * omega);
        else if (seg == 2) val = sinf((float)gh * omega);
        else               val = cosf((float)gh * omega);
        pe[(size_t)n * Dd + d] = val;
    }
}

__global__ __launch_bounds__(256) void fused_kernel(
        const float* __restrict__ x,
        const float* __restrict__ ln1_g, const float* __restrict__ ln1_b,
        const float* __restrict__ b_proj,
        const float* __restrict__ ln2_g, const float* __restrict__ ln2_b,
        const signed char* __restrict__ wq,
        const float* __restrict__ sw_inv_p,
        const float* __restrict__ pe,
        float* __restrict__ out)
{
    __shared__ float s_patch[16][148];
    __shared__ __align__(32) int s_afrag[3 * 32 * 8];
    __shared__ float s_rowscale[16];
    __shared__ float s_red[16][16];
    __shared__ float s_red2[16][16];
    __shared__ float s_wsum[8][16];
    __shared__ float s_wsq[8][16];
    __shared__ float s_mu2[16];
    __shared__ float s_rs2[16];

    const int tid = threadIdx.x;
    const int ty = tid >> 4, tx = tid & 15;
    const int tok0 = blockIdx.x * 16;

    for (int i = tid; i < 3 * 32 * 8; i += 256) s_afrag[i] = 0;

    {
        const int token = tok0 + ty;
        const int b = token >> 10, n = token & 1023;
        const int gh = n >> 5, gw = n & 31;
        float sum = 0.0f, sq = 0.0f;
        for (int e = tx; e < PD; e += 16) {
            int p1 = e / 21, rem = e - p1 * 21;
            int p2 = rem / 3, c = rem - p2 * 3;
            size_t addr = (((size_t)(b * Cc + c) * Hh) + (gh * Pp + p1)) * Ww + (gw * Pp + p2);
            float v = x[addr];
            s_patch[ty][e] = v;
            sum += v; sq += v * v;
        }
        s_red[ty][tx] = sum; s_red2[ty][tx] = sq;
        __syncthreads();
        for (int off = 8; off > 0; off >>= 1) {
            if (tx < off) {
                s_red[ty][tx]  += s_red[ty][tx + off];
                s_red2[ty][tx] += s_red2[ty][tx + off];
            }
            __syncthreads();
        }
        const float inv147 = 1.0f / 147.0f;
        float mu   = s_red[ty][0] * inv147;
        float var  = s_red2[ty][0] * inv147 - mu * mu;
        float rstd = rsqrtf(var + EPS);
        __syncthreads();
        float amax = 0.0f;
        for (int e = tx; e < PD; e += 16) {
            float v = (s_patch[ty][e] - mu) * rstd * ln1_g[e] + ln1_b[e];
            s_patch[ty][e] = v;
            amax = fmaxf(amax, fabsf(v));
        }
        s_red[ty][tx] = amax;
        __syncthreads();
        for (int off = 8; off > 0; off >>= 1) {
            if (tx < off) s_red[ty][tx] = fmaxf(s_red[ty][tx], s_red[ty][tx + off]);
            __syncthreads();
        }
        float am = fmaxf(s_red[ty][0], 1e-5f);
        float sx = 127.0f / am;
        if (tx == 0) s_rowscale[ty] = (am * (1.0f / 127.0f)) * sw_inv_p[0];
        for (int e = tx; e < PD; e += 16) {
            float v = s_patch[ty][e];
            int q = (int)rintf(v * sx);
            q = q > 127 ? 127 : (q < -128 ? -128 : q);
            int kb = e >> 6, k = e & 63;
            int j = k >> 4, r = k & 15, h2 = j & 1, w2 = r >> 2, byt = r & 3;
            int p = 4 * (j >> 1) + w2, lane2 = ty + 16 * h2;
            ((signed char*)s_afrag)[(((kb * 32 + lane2) * 8) + p) * 4 + byt] = (signed char)q;
        }
    }
    __syncthreads();

    const int wave = tid >> 5, lane = tid & 31;
    const int h = lane >> 4, ln = lane & 15;

    v8i a0 = *(const v8i*)&s_afrag[(0 * 32 + lane) * 8];
    v8i a1 = *(const v8i*)&s_afrag[(1 * 32 + lane) * 8];
    v8i a2 = *(const v8i*)&s_afrag[(2 * 32 + lane) * 8];

    float rs8[8];
#pragma unroll
    for (int r = 0; r < 8; ++r) rs8[r] = s_rowscale[8 * h + r];

    float accf[8][8];
    float psum[8], psq[8];
#pragma unroll
    for (int r = 0; r < 8; ++r) { psum[r] = 0.0f; psq[r] = 0.0f; }

#pragma unroll
    for (int i = 0; i < 8; ++i) {
        const int nt = wave * 8 + i;
        const signed char* bp = wq + ((size_t)(nt * 3) * 32 + lane) * 32;
        v8i b0 = *(const v8i*)(bp);
        v8i b1 = *(const v8i*)(bp + 1024);
        v8i b2 = *(const v8i*)(bp + 2048);
        v8i acc = (v8i){0, 0, 0, 0, 0, 0, 0, 0};
        acc = __builtin_amdgcn_wmma_i32_16x16x64_iu8(true, a0, true, b0, acc, false, false);
        acc = __builtin_amdgcn_wmma_i32_16x16x64_iu8(true, a1, true, b1, acc, false, false);
        acc = __builtin_amdgcn_wmma_i32_16x16x64_iu8(true, a2, true, b2, acc, false, false);
        const int col = nt * 16 + ln;
        const float bias = b_proj[col];
#pragma unroll
        for (int r = 0; r < 8; ++r) {
            float f = (float)acc[r] * rs8[r] + bias;
            accf[i][r] = f;
            psum[r] += f; psq[r] += f * f;
        }
    }

#pragma unroll
    for (int m2 = 1; m2 < 16; m2 <<= 1) {
#pragma unroll
        for (int r = 0; r < 8; ++r) {
            psum[r] += __shfl_xor(psum[r], m2, 32);
            psq[r]  += __shfl_xor(psq[r],  m2, 32);
        }
    }
    if (ln == 0) {
#pragma unroll
        for (int r = 0; r < 8; ++r) {
            s_wsum[wave][8 * h + r] = psum[r];
            s_wsq [wave][8 * h + r] = psq[r];
        }
    }
    __syncthreads();
    if (tid < 16) {
        float su = 0.0f, s2 = 0.0f;
        for (int w = 0; w < 8; ++w) { su += s_wsum[w][tid]; s2 += s_wsq[w][tid]; }
        float mu  = su * (1.0f / (float)Dd);
        float var = s2 * (1.0f / (float)Dd) - mu * mu;
        s_mu2[tid] = mu;
        s_rs2[tid] = rsqrtf(var + EPS);
    }
    __syncthreads();

#pragma unroll
    for (int i = 0; i < 8; ++i) {
        const int nt = wave * 8 + i;
        const int col = nt * 16 + ln;
        const float g2 = ln2_g[col], bb2 = ln2_b[col];
#pragma unroll
        for (int r = 0; r < 8; ++r) {
            const int m = 8 * h + r;
            const int token = tok0 + m;
            const int n = token & 1023;
            float v = (accf[i][r] - s_mu2[m]) * s_rs2[m] * g2 + bb2
                      + pe[(size_t)n * Dd + col];
            out[(size_t)token * Dd + col] = v;
        }
    }
}

extern "C" void kernel_launch(void* const* d_in, const int* in_sizes, int n_in,
                              void* d_out, int out_size, void* d_ws, size_t ws_size,
                              hipStream_t stream) {
    (void)in_sizes; (void)n_in; (void)out_size; (void)ws_size;
    const float* x      = (const float*)d_in[0];
    const float* ln1_g  = (const float*)d_in[1];
    const float* ln1_b  = (const float*)d_in[2];
    const float* W_proj = (const float*)d_in[3];
    const float* b_proj = (const float*)d_in[4];
    const float* ln2_g  = (const float*)d_in[5];
    const float* ln2_b  = (const float*)d_in[6];
    float* out = (float*)d_out;

    char* ws = (char*)d_ws;
    float*       sw_inv = (float*)ws;
    signed char* wq     = (signed char*)(ws + 256);
    float*       pe     = (float*)(ws + 256 + Dd * KPAD);

    wmean_kernel  <<<1,    256, 0, stream>>>(W_proj, sw_inv);
    wquant_kernel <<<768,  256, 0, stream>>>(W_proj, sw_inv, wq);
    posemb_kernel <<<1024, 256, 0, stream>>>(pe);
    fused_kernel  <<<(Bn * NTOK) / 16, 256, 0, stream>>>(
        x, ln1_g, ln1_b, b_proj, ln2_g, ln2_b, wq, sw_inv, pe, out);
}
